// HierarchicalAssociativeMemory_51376398795451
// MI455X (gfx1250) — matrix operand order repaired and checked by reading, not compiled, not run on hardware
//
#include <hip/hip_runtime.h>
#include <math.h>

typedef __attribute__((ext_vector_type(16))) _Float16 v16h;
typedef __attribute__((ext_vector_type(8)))  _Float16 v8h;
typedef __attribute__((ext_vector_type(8)))  float    v8f;

#define BATCHN 131072

// ---- workspace layout ----
// float region (element offsets into (float*)d_ws)
#define WS_K0   0       // k0 = keys0@k_w0.T + k_b0   [64][64]
#define WS_K1   4096    // k1 = keys1@k_w1.T + k_b1   [32][128]
#define WS_C0   8192    // composed score bias lvl0   [256]
#define WS_C1   8448    // composed score bias lvl1   [128]
// f16 region starts at this byte offset (64B aligned)
#define WS_H_BYTE 34816
// element offsets into (_Float16*)((char*)d_ws + WS_H_BYTE); all [N][K] row-major
#define H_M0   0        // [256][64]  fused q_w0 ∘ k0, scaled 1/4
#define H_M1   16384    // [128][128] fused q_w1 ∘ k1, scaled 1/sqrt(32)
#define H_V0   32768    // [64][64]   vals0 ∘ vo_w0.T  (N=e, K=m)
#define H_V1   36864    // [128][32]  vals1 ∘ vo_w1.T
#define H_UP   40960    // [128][64]  up_w0 (already [out][in])
#define H_DN   49152    // [64][128]  dn_w0
#define H_G11  57344    // [32][128]  g1_w1
#define H_G10  61440    // [16][64]   g1_w0

// ===================== prep kernels (tiny) =====================
__global__ void prep_k(const float* __restrict__ keys0, const float* __restrict__ k_w0,
                       const float* __restrict__ k_b0,
                       const float* __restrict__ keys1, const float* __restrict__ k_w1,
                       const float* __restrict__ k_b1, float* __restrict__ wsF) {
  int tid = blockIdx.x * blockDim.x + threadIdx.x;
  if (tid < 4096) {                       // k0[m][e], m,e < 64
    int m = tid >> 6, e = tid & 63;
    float acc = k_b0[e];
    for (int d = 0; d < 64; ++d) acc += keys0[m * 64 + d] * k_w0[e * 64 + d];
    wsF[WS_K0 + m * 64 + e] = acc;
  } else if (tid < 8192) {                // k1[m][e], m<32, e<128
    int t = tid - 4096;
    int m = t >> 7, e = t & 127;
    float acc = k_b1[e];
    for (int d = 0; d < 128; ++d) acc += keys1[m * 128 + d] * k_w1[e * 128 + d];
    wsF[WS_K1 + m * 128 + e] = acc;
  }
}

__global__ void prep_fused(const float* __restrict__ wsF, _Float16* __restrict__ wsH,
                           float* __restrict__ wsFo,
                           const float* __restrict__ q_w0, const float* __restrict__ q_b0,
                           const float* __restrict__ q_w1, const float* __restrict__ q_b1,
                           const float* __restrict__ vals0, const float* __restrict__ vo_w0,
                           const float* __restrict__ vals1, const float* __restrict__ vo_w1) {
  const float s0 = 0.25f;                  // 1/sqrt(16)
  const float s1 = 0.17677669529663687f;   // 1/sqrt(32)
  const float* k0 = wsF + WS_K0;
  const float* k1 = wsF + WS_K1;
  int tid = blockIdx.x * blockDim.x + threadIdx.x;
  if (tid < 16384) {                       // M0t[n=h*64+m][k], k<64
    int n = tid >> 6, k = tid & 63;
    int h = n >> 6, m = n & 63;
    float acc = 0.f;
    for (int j = 0; j < 16; ++j)
      acc += q_w0[(h * 16 + j) * 64 + k] * k0[m * 64 + h * 16 + j];
    wsH[H_M0 + n * 64 + k] = (_Float16)(acc * s0);
  } else if (tid < 32768) {                // M1t[n=h*32+m][k], k<128
    int i = tid - 16384;
    int n = i >> 7, k = i & 127;
    int h = n >> 5, m = n & 31;
    float acc = 0.f;
    for (int j = 0; j < 32; ++j)
      acc += q_w1[(h * 32 + j) * 128 + k] * k1[m * 128 + h * 32 + j];
    wsH[H_M1 + n * 128 + k] = (_Float16)(acc * s1);
  } else if (tid < 36864) {                // V0t[e][m] = sum_d vo_w0[e,d]*vals0[m,d]
    int i = tid - 32768;
    int e = i >> 6, m = i & 63;
    float acc = 0.f;
    for (int d = 0; d < 64; ++d) acc += vo_w0[e * 64 + d] * vals0[m * 64 + d];
    wsH[H_V0 + e * 64 + m] = (_Float16)acc;
  } else if (tid < 40960) {                // V1t[e][m], e<128, m<32
    int i = tid - 36864;
    int e = i >> 5, m = i & 31;
    float acc = 0.f;
    for (int d = 0; d < 128; ++d) acc += vo_w1[e * 128 + d] * vals1[m * 128 + d];
    wsH[H_V1 + e * 32 + m] = (_Float16)acc;
  } else if (tid < 41216) {                // C0[n=h*64+m]
    int n = tid - 40960;
    int h = n >> 6, m = n & 63;
    float acc = 0.f;
    for (int j = 0; j < 16; ++j) acc += q_b0[h * 16 + j] * k0[m * 64 + h * 16 + j];
    wsFo[WS_C0 + n] = acc * s0;
  } else if (tid < 41344) {                // C1[n=h*32+m]
    int n = tid - 41216;
    int h = n >> 5, m = n & 31;
    float acc = 0.f;
    for (int j = 0; j < 32; ++j) acc += q_b1[h * 32 + j] * k1[m * 128 + h * 32 + j];
    wsFo[WS_C1 + n] = acc * s1;
  }
}

__global__ void prep_cvt(_Float16* __restrict__ wsH,
                         const float* __restrict__ up_w0, const float* __restrict__ dn_w0,
                         const float* __restrict__ g1_w1, const float* __restrict__ g1_w0) {
  int tid = blockIdx.x * blockDim.x + threadIdx.x;
  if (tid < 8192)        wsH[H_UP  + tid]          = (_Float16)up_w0[tid];           // [128][64]
  else if (tid < 16384)  wsH[H_DN  + tid - 8192]   = (_Float16)dn_w0[tid - 8192];    // [64][128]
  else if (tid < 20480)  wsH[H_G11 + tid - 16384]  = (_Float16)g1_w1[tid - 16384];   // [32][128]
  else if (tid < 21504)  wsH[H_G10 + tid - 20480]  = (_Float16)g1_w0[tid - 20480];   // [16][64]
}

// ===================== main kernel =====================
__device__ __forceinline__ void wsync() {
  asm volatile("s_wait_dscnt 0" ::: "memory");
}

// A operand (16xK activation tile in LDS, row-major stride K), f16 16x16x32 layout
__device__ __forceinline__ v16h load_a(const _Float16* A, int K, int kb, int lane) {
  int g = (lane >> 4) & 1, r = lane & 15;
  const _Float16* p = A + r * K + kb + g * 8;
  v8h c0 = *(const v8h*)p;          // K = kb+8g .. +7      (halves t=0..7)
  v8h c1 = *(const v8h*)(p + 16);   // K = kb+16+8g .. +7   (halves t=8..15)
  v16h a;
#pragma unroll
  for (int i = 0; i < 8; ++i) { a[i] = c0[i]; a[i + 8] = c1[i]; }
  return a;
}

// B operand from global weights stored [N][K]; lane n=lane%16 col, K=8*(lane/16)+t (t<8), 16+8*(lane/16)+t-8 (t>=8)
__device__ __forceinline__ v16h load_b(const _Float16* Wt, int K, int nt, int kb, int lane) {
  int g = (lane >> 4) & 1, n = lane & 15;
  const _Float16* p = Wt + (size_t)(nt * 16 + n) * K + kb + g * 8;
  v8h c0 = *(const v8h*)p;
  v8h c1 = *(const v8h*)(p + 16);
  v16h b;
#pragma unroll
  for (int i = 0; i < 8; ++i) { b[i] = c0[i]; b[i + 8] = c1[i]; }
  return b;
}

// Out[16][ncol] (LDS, f32) = A[16][K] @ Wt.T + bias    (Wt stored [ncol][K])
__device__ __forceinline__ void gemm(const _Float16* A, int K, const _Float16* Wt,
                                     const float* bias, float* Out, int ncol, int lane) {
  int g = (lane >> 4) & 1, n = lane & 15;
  for (int nt = 0; nt < (ncol >> 4); ++nt) {
    v8f acc = {};
    for (int kb = 0; kb < K; kb += 32) {
      v16h a = load_a(A, K, kb, lane);
      v16h b = load_b(Wt, K, nt, kb, lane);
      acc = __builtin_amdgcn_wmma_f32_16x16x32_f16(false, a, false, b, (short)0, acc,
                                                   false, false);
    }
    float bv = bias[nt * 16 + n];
#pragma unroll
    for (int r = 0; r < 8; ++r)
      Out[(r + 8 * g) * ncol + nt * 16 + n] = acc[r] + bv;
  }
}

__global__ __launch_bounds__(64) void ham_main(
    const float* __restrict__ x, const _Float16* __restrict__ wsH,
    const float* __restrict__ wsF, const float* __restrict__ vo_b0,
    const float* __restrict__ up_b0, const float* __restrict__ vo_b1,
    const float* __restrict__ g1_b1, const float* __restrict__ g2_w1,
    const float* __restrict__ g2_b1, const float* __restrict__ dn_b0,
    const float* __restrict__ g1_b0, const float* __restrict__ g2_w0,
    const float* __restrict__ g2_b0, float* __restrict__ out) {
  __shared__ _Float16 sA[2][16 * 128];
  __shared__ float    sS[2][16 * 128];
  __shared__ float    sT[2][16 * 128];
  __shared__ float    sW[2][16 * 32];
  __shared__ float    sG[2][16];

  const int wave = threadIdx.x >> 5;
  const int lane = threadIdx.x & 31;
  const int tile = blockIdx.x * 2 + wave;
  const size_t row0 = (size_t)tile * 16;

  _Float16* A = sA[wave];
  float *S = sS[wave], *T = sT[wave], *W = sW[wave], *G = sG[wave];
  const _Float16 *M0t = wsH + H_M0, *M1t = wsH + H_M1, *V0t = wsH + H_V0,
                 *V1t = wsH + H_V1, *UPt = wsH + H_UP, *DNt = wsH + H_DN,
                 *G11t = wsH + H_G11, *G10t = wsH + H_G10;
  const float *C0 = wsF + WS_C0, *C1 = wsF + WS_C1;
  const float* xr = x + row0 * 64;

  // x -> A (f16, 16x64)
  for (int i = lane; i < 1024; i += 32) A[i] = (_Float16)xr[i];
  // zero head-avg weight accumulator (T as 16x64)
  for (int i = lane; i < 1024; i += 32) T[i] = 0.f;
  wsync();

  // ---- level-0 read: scores per head -> softmax -> averaged weights ----
  for (int h = 0; h < 4; ++h) {
    gemm(A, 64, M0t + h * 64 * 64, C0 + h * 64, S, 64, lane);
    wsync();
    if (lane < 16) {
      float* sr = S + lane * 64;
      float mx = -1e30f;
      for (int c = 0; c < 64; ++c) mx = fmaxf(mx, sr[c]);
      float s = 0.f;
      for (int c = 0; c < 64; ++c) { float e = __expf(sr[c] - mx); sr[c] = e; s += e; }
      float inv = 0.25f / s;
      float* wr = T + lane * 64;
      for (int c = 0; c < 64; ++c) wr[c] += sr[c] * inv;
    }
    wsync();
  }
  for (int i = lane; i < 1024; i += 32) A[i] = (_Float16)T[i];  // w0 -> A (16x64)
  wsync();
  gemm(A, 64, V0t, vo_b0, S, 64, lane);                         // r0 (16x64)
  wsync();
  for (int i = lane; i < 1024; i += 32) A[i] = (_Float16)S[i];
  wsync();
  gemm(A, 64, UPt, up_b0, S, 128, lane);                        // cur1 f32 kept in S
  wsync();
  for (int i = lane; i < 2048; i += 32) A[i] = (_Float16)S[i];  // cur1 -> A (16x128)
  wsync();

  // ---- level-1 read ----
  gemm(A, 128, M1t, C1, T, 128, lane);                          // scores1 (16x128)
  wsync();
  for (int i = lane; i < 512; i += 32) W[i] = 0.f;              // w1 accumulator
  wsync();
  if (lane < 16) {
    for (int h = 0; h < 4; ++h) {
      float* sr = T + lane * 128 + h * 32;
      float mx = -1e30f;
      for (int c = 0; c < 32; ++c) mx = fmaxf(mx, sr[c]);
      float s = 0.f;
      for (int c = 0; c < 32; ++c) { float e = __expf(sr[c] - mx); sr[c] = e; s += e; }
      float inv = 0.25f / s;
      float* wr = W + lane * 32;
      for (int c = 0; c < 32; ++c) wr[c] += sr[c] * inv;
    }
  }
  wsync();
  for (int i = lane; i < 512; i += 32) A[i] = (_Float16)W[i];   // w1 -> A (16x32)
  wsync();
  gemm(A, 32, V1t, vo_b1, T, 128, lane);                        // ho1 f32 in T
  wsync();
  for (int i = lane; i < 2048; i += 32) A[i] = (_Float16)T[i];  // ho1 -> A
  wsync();
  gemm(A, 128, G11t, g1_b1, W, 32, lane);                       // gate1 hidden (16x32)
  wsync();
  if (lane < 16) {
    float acc = g2_b1[0];
    const float* hr = W + lane * 32;
    for (int c = 0; c < 32; ++c) {
      float v = hr[c];
      float ge = 0.5f * v * (1.f + erff(v * 0.70710678118654752f));
      acc += ge * g2_w1[c];
    }
    G[lane] = 1.f / (1.f + __expf(-acc));
  }
  wsync();
  // out1 = g*ho1 + (1-g)*cur1  (T=ho1, S=cur1) -> A (f16)
  for (int i = lane; i < 2048; i += 32) {
    float gg = G[i >> 7];
    A[i] = (_Float16)(gg * T[i] + (1.f - gg) * S[i]);
  }
  wsync();

  // ---- down path ----
  gemm(A, 128, DNt, dn_b0, S, 64, lane);                        // cur f32 in S (16x64)
  wsync();
  for (int i = lane; i < 1024; i += 32) A[i] = (_Float16)S[i];
  wsync();
  gemm(A, 64, G10t, g1_b0, W, 16, lane);                        // gate0 hidden (16x16)
  wsync();
  if (lane < 16) {
    float acc = g2_b0[0];
    const float* hr = W + lane * 16;
    for (int c = 0; c < 16; ++c) {
      float v = hr[c];
      float ge = 0.5f * v * (1.f + erff(v * 0.70710678118654752f));
      acc += ge * g2_w0[c];
    }
    G[lane] = 1.f / (1.f + __expf(-acc));
  }
  wsync();
  float* orow = out + row0 * 64;
  for (int i = lane; i < 1024; i += 32) {
    float gg = G[i >> 6];
    orow[i] = gg * S[i] + (1.f - gg) * xr[i];
  }
}

// ===================== launcher =====================
// d_in order (setup_inputs dict flattened: x, then params in insertion order):
// 0:x 1:keys0 2:vals0 3:q_w0 4:q_b0 5:k_w0 6:k_b0 7:vo_w0 8:vo_b0 9:g1_w0 10:g1_b0
// 11:g2_w0 12:g2_b0 13..24: level1 same order, 25..36: level2 (unused),
// 37:up_w0 38:up_b0 39:dn_w0 40:dn_b0 41..44: up/dn level1 (unused)
extern "C" void kernel_launch(void* const* d_in, const int* in_sizes, int n_in,
                              void* d_out, int out_size, void* d_ws, size_t ws_size,
                              hipStream_t stream) {
  const float* x     = (const float*)d_in[0];
  const float* keys0 = (const float*)d_in[1];
  const float* vals0 = (const float*)d_in[2];
  const float* q_w0  = (const float*)d_in[3];
  const float* q_b0  = (const float*)d_in[4];
  const float* k_w0  = (const float*)d_in[5];
  const float* k_b0  = (const float*)d_in[6];
  const float* vo_w0 = (const float*)d_in[7];
  const float* vo_b0 = (const float*)d_in[8];
  const float* g1_w0 = (const float*)d_in[9];
  const float* g1_b0 = (const float*)d_in[10];
  const float* g2_w0 = (const float*)d_in[11];
  const float* g2_b0 = (const float*)d_in[12];
  const float* keys1 = (const float*)d_in[13];
  const float* vals1 = (const float*)d_in[14];
  const float* q_w1  = (const float*)d_in[15];
  const float* q_b1  = (const float*)d_in[16];
  const float* k_w1  = (const float*)d_in[17];
  const float* k_b1  = (const float*)d_in[18];
  const float* vo_w1 = (const float*)d_in[19];
  const float* vo_b1 = (const float*)d_in[20];
  const float* g1_w1 = (const float*)d_in[21];
  const float* g1_b1 = (const float*)d_in[22];
  const float* g2_w1 = (const float*)d_in[23];
  const float* g2_b1 = (const float*)d_in[24];
  const float* up_w0 = (const float*)d_in[37];
  const float* up_b0 = (const float*)d_in[38];
  const float* dn_w0 = (const float*)d_in[39];
  const float* dn_b0 = (const float*)d_in[40];

  float* wsF = (float*)d_ws;
  _Float16* wsH = (_Float16*)((char*)d_ws + WS_H_BYTE);

  prep_k<<<(8192 + 255) / 256, 256, 0, stream>>>(keys0, k_w0, k_b0, keys1, k_w1, k_b1, wsF);
  prep_fused<<<(41344 + 255) / 256, 256, 0, stream>>>(wsF, wsH, wsF, q_w0, q_b0, q_w1, q_b1,
                                                      vals0, vo_w0, vals1, vo_w1);
  prep_cvt<<<(21504 + 255) / 256, 256, 0, stream>>>(wsH, up_w0, dn_w0, g1_w1, g1_w0);

  ham_main<<<BATCHN / 32, 64, 0, stream>>>(x, wsH, wsF, vo_b0, up_b0, vo_b1, g1_b1, g2_w1,
                                           g2_b1, dn_b0, g1_b0, g2_w0, g2_b0,
                                           (float*)d_out);
}
